// mGNN_4715874091923
// MI455X (gfx1250) — compile-verified, matrix operand order checked by reading, not run on hardware
//
#include <hip/hip_runtime.h>
#include <hip/hip_bf16.h>
#include <math.h>

typedef __attribute__((ext_vector_type(16))) __bf16 v16bf;
typedef __attribute__((ext_vector_type(8)))  __bf16 v8bf;
typedef __attribute__((ext_vector_type(8)))  float  v8f;

#define CDIV(a,b) (((a)+(b)-1)/(b))

// ---------------------------------------------------------------- helpers
__device__ __forceinline__ float hswishf(float x) {
  float r = fminf(fmaxf(x + 3.0f, 0.0f), 6.0f);
  return x * r * (1.0f / 6.0f);
}
__device__ __forceinline__ float sigmf(float x) { return 1.0f / (1.0f + expf(-x)); }

// monotone float<->uint encoding for atomicMax on signed floats
__device__ __forceinline__ unsigned f2ord(float f) {
  unsigned u = __float_as_uint(f);
  return (u & 0x80000000u) ? ~u : (u | 0x80000000u);
}
__device__ __forceinline__ float ord2f(unsigned u) {
  unsigned v = (u & 0x80000000u) ? (u & 0x7FFFFFFFu) : ~u;
  return __uint_as_float(v);
}
__device__ __forceinline__ float dec_cell(unsigned u) {
  float f = ord2f(u);
  return (f - f == 0.0f) ? f : 0.0f;   // non-finite (untouched cell) -> 0
}

// ------------------------------------------------ WMMA bf16 fragment loads
// A 16x32 bf16 (ISA 7.12.2): lane<16: row=lane, K={kb..kb+7, kb+16..kb+23}, kb=(lane>>4)*8.
// Per-lane data = two contiguous 16B chunks -> two b128 loads. UNGUARDED.
__device__ __forceinline__ v16bf load_a_rm(const __bf16* A, int lda, int k0) {
  int lane = threadIdx.x & 31;
  const __bf16* p = A + (long)(lane & 15) * lda + k0 + ((lane >> 4) << 3);
  v8bf lo = *(const v8bf*)p;
  v8bf hi = *(const v8bf*)(p + 16);
  v16bf a;
#pragma unroll
  for (int i = 0; i < 8; ++i) { a[i] = lo[i]; a[i + 8] = hi[i]; }
  return a;
}
// B fragment from pre-swizzled storage: tile = [32 lanes][16 bf16] contiguous (1KB/tile)
__device__ __forceinline__ v16bf load_b_staged(const __bf16* S, long tile) {
  const __bf16* p = S + tile * 512 + ((threadIdx.x & 31) << 4);
  v8bf lo = *(const v8bf*)p;
  v8bf hi = *(const v8bf*)(p + 8);
  v16bf b;
#pragma unroll
  for (int i = 0; i < 8; ++i) { b[i] = lo[i]; b[i + 8] = hi[i]; }
  return b;
}
#define WMMA_BF16(a,b,c) __builtin_amdgcn_wmma_f32_16x16x32_bf16(false,(a),false,(b),(short)0,(c),false,false)

// ---------------------------------------------------------------- utility
__global__ void k_zero_f32(float* p, long n) {
  long i = (long)blockIdx.x * blockDim.x + threadIdx.x;
  if (i < n) p[i] = 0.0f;
}
__global__ void k_fill_u32(unsigned* p, unsigned v, long n) {
  long i = (long)blockIdx.x * blockDim.x + threadIdx.x;
  if (i < n) p[i] = v;
}
__global__ void k_x_to_hist(const float* x, __bf16* hist, int n) {
  long i = (long)blockIdx.x * blockDim.x + threadIdx.x;
  if (i < (long)n * 4) hist[(i >> 2) * 644 + 640 + (i & 3)] = (__bf16)x[i];
}

// Stage a K x N row-major f32 weight into fragment layout, K zero-padded to KT*32.
// S[(kt*NT + nt)*512 + lane*16 + ii]
__global__ void k_stage_kn(const float* __restrict__ W, __bf16* __restrict__ S,
                           int K, int Ncols, long total) {
  long i = (long)blockIdx.x * blockDim.x + threadIdx.x;
  if (i >= total) return;
  int ii   = (int)(i & 15);
  int lane = (int)((i >> 4) & 31);
  long tile = i >> 9;
  int NT = Ncols / 16;
  int nt = (int)(tile % NT), kt = (int)(tile / NT);
  int col = nt * 16 + (lane & 15);
  int k = kt * 32 + ((lane >> 4) << 3) + (ii < 8 ? ii : ii + 8);
  S[i] = (k < K) ? (__bf16)W[(long)k * Ncols + col] : (__bf16)0.0f;
}

// Stage grouped node weight W[g][cgo][Kg] as B[k][n] = W[n/cgo][n%cgo][k], K padded to KT*32.
__global__ void k_stage_nodeW(const float* __restrict__ W, __bf16* __restrict__ S,
                              int Kg, int cgo, int cout, long total) {
  long i = (long)blockIdx.x * blockDim.x + threadIdx.x;
  if (i >= total) return;
  int ii   = (int)(i & 15);
  int lane = (int)((i >> 4) & 31);
  long tile = i >> 9;
  int NT = cout / 16;
  int nt = (int)(tile % NT), kt = (int)(tile / NT);
  int col = nt * 16 + (lane & 15);
  int gi = col / cgo, cl = col - gi * cgo;
  int k = kt * 32 + ((lane >> 4) << 3) + (ii < 8 ? ii : ii + 8);
  S[i] = (k < Kg) ? (__bf16)W[((long)gi * cgo + cl) * Kg + k] : (__bf16)0.0f;
}

// --------------------------------------------------------- node linear ops
// block0: cin=4, cout=64 (dense + bias) — trivial scalar GEMM
__global__ void k_node0(const float* x, const float* w, const float* b, float* Y, int n) {
  long i = (long)blockIdx.x * blockDim.x + threadIdx.x;
  if (i >= (long)n * 64) return;
  int c = (int)(i & 63); long nn = i >> 6;
  float s = b[c];
#pragma unroll
  for (int k = 0; k < 4; ++k) s += x[nn * 4 + k] * w[k * 64 + c];
  Y[i] = s;
}

// grouped node GEMM via WMMA; A overrun beyond Kg is harmless (B pad zero, hist zero-init)
__global__ __launch_bounds__(32) void k_node_gemm_g(
    const __bf16* __restrict__ hist, int hcol,
    const __bf16* __restrict__ SW, int cout, int Kg, int cgo, int KT,
    float* __restrict__ Y)
{
  int m0 = blockIdx.x * 16;
  int nt = blockIdx.y;
  int n0 = nt * 16;
  int NT = cout / 16;
  int gi = n0 / cgo;
  int lane = threadIdx.x & 31;
  const __bf16* Ab = hist + (long)m0 * 644 + hcol + gi * Kg;
  v8f acc = {};
  for (int kt = 0; kt < KT; ++kt) {
    v16bf a = load_a_rm(Ab, 644, kt * 32);
    v16bf b = load_b_staged(SW, (long)kt * NT + nt);
    acc = WMMA_BF16(a, b, acc);
  }
  int col = n0 + (lane & 15);
  int rb  = (lane >> 4) * 8;
#pragma unroll
  for (int r = 0; r < 8; ++r)
    Y[(long)(m0 + rb + r) * cout + col] = acc[r];
}

// per-column mean / inv-std (biased var, eps=1e-5)
__global__ void k_colstats(const float* __restrict__ Y, int rows, int cols,
                           float* mean, float* istd) {
  int c = blockIdx.x;
  __shared__ float s1[256], s2[256];
  float a = 0.f, b = 0.f;
  for (int n = threadIdx.x; n < rows; n += 256) {
    float v = Y[(long)n * cols + c];
    a += v; b += v * v;
  }
  s1[threadIdx.x] = a; s2[threadIdx.x] = b;
  __syncthreads();
  for (int s = 128; s > 0; s >>= 1) {
    if (threadIdx.x < s) { s1[threadIdx.x] += s1[threadIdx.x + s]; s2[threadIdx.x] += s2[threadIdx.x + s]; }
    __syncthreads();
  }
  if (threadIdx.x == 0) {
    float m = s1[0] / rows;
    float v = s2[0] / rows - m * m;
    mean[c] = m;
    istd[c] = rsqrtf(v + 1e-5f);
  }
}

// BN + hswish + (grouped) channel shuffle -> H (f32 + bf16)
__global__ void k_bn_apply_node(const float* __restrict__ Y, const float* mean, const float* istd,
                                const float* gam, const float* bet,
                                int cout, int g, int cgo,
                                float* __restrict__ Hf, __bf16* __restrict__ Hb, long total) {
  long i = (long)blockIdx.x * blockDim.x + threadIdx.x;
  if (i >= total) return;
  int cp = (int)(i % cout); long n = i / cout;
  float v = (Y[i] - mean[cp]) * istd[cp] * gam[cp] + bet[cp];
  v = hswishf(v);
  int cs = (g == 1) ? cp : (cp % cgo) * g + cp / cgo;
  Hf[n * cout + cs] = v;
  Hb[n * cout + cs] = (__bf16)v;
}

// out = hswish(h + agg) -> hist columns (blocks 0..3) or XOUT (block 4)
__global__ void k_finalize(const float* __restrict__ Hf, const float* __restrict__ agg, int cout,
                           __bf16* hist, int hcol, float* xout, long total) {
  long i = (long)blockIdx.x * blockDim.x + threadIdx.x;
  if (i >= total) return;
  int c = (int)(i % cout); long n = i / cout;
  float v = hswishf(Hf[i] + agg[i]);
  if (hist) hist[n * 644 + hcol + c] = (__bf16)v;
  if (xout) xout[i] = v;
}

// ------------------------------------------------------------ edge kernel
// 16 edges x COUT per workgroup, 4 waves, WMMA bf16 tiles, staged weights.
template<int COUT>
__global__ __launch_bounds__(128) void k_edge(
    const __bf16* __restrict__ H,
    const int* __restrict__ esrc, const int* __restrict__ edst, int E, int Nn,
    const __bf16* __restrict__ SWsc, const float* __restrict__ bs,
    const __bf16* __restrict__ SW1, const float* __restrict__ b1,
    const __bf16* __restrict__ SW2, const float* __restrict__ b2,
    float* __restrict__ agg)
{
  constexpr int CQ  = COUT / 4;
  constexpr int CQP = ((CQ + 31) / 32) * 32;   // padded K for msg GEMM
  constexpr int NT  = COUT / 16;
  constexpr int NT1 = CQ / 16;
  constexpr int KT  = COUT / 32;
  constexpr int KT2 = CQP / 32;

  __shared__ __align__(16) __bf16 sh_d [16 * COUT];
  __shared__ __align__(16) __bf16 sh_xj[16 * COUT];
  __shared__ __align__(16) __bf16 sh_t1[16 * CQP];
  __shared__ float sh_sc[16 * COUT];
  __shared__ int sh_si[16], sh_di[16], sh_ok[16];

  const int tid  = threadIdx.x;
  const int wave = tid >> 5;
  const int lane = tid & 31;
  const long e0  = (long)blockIdx.x * 16;
  const long E2  = (long)E + Nn;

  if (tid < 16) {
    long e = e0 + tid;
    int ok = (e < E2);
    long ec = ok ? e : 0;
    sh_si[tid] = (ec < E) ? esrc[ec] : (int)(ec - E);
    sh_di[tid] = (ec < E) ? edst[ec] : (int)(ec - E);
    sh_ok[tid] = ok;
  }
  // zero t1 pad region (whole buffer; valid cols overwritten later)
  {
    v8bf z = {};
    for (int idx = tid; idx < 16 * CQP / 8; idx += 128)
      *(v8bf*)(sh_t1 + idx * 8) = z;
  }
  __syncthreads();

  // vectorized gather of xj / xi rows; d = xj - xi
  for (int idx = tid; idx < 16 * COUT / 8; idx += 128) {
    int r  = idx / (COUT / 8);
    int cc = (idx - r * (COUT / 8)) * 8;
    v8bf xj = *(const v8bf*)(H + (long)sh_si[r] * COUT + cc);
    v8bf xi = *(const v8bf*)(H + (long)sh_di[r] * COUT + cc);
    v8bf d;
#pragma unroll
    for (int q = 0; q < 8; ++q) d[q] = (__bf16)((float)xj[q] - (float)xi[q]);
    *(v8bf*)(sh_xj + r * COUT + cc) = xj;
    *(v8bf*)(sh_d  + r * COUT + cc) = d;
  }
  __syncthreads();

  const int col0 = lane & 15;
  const int rb   = (lane >> 4) * 8;

  // score = sigmoid((xj-xi)@Wsc + bs)
  for (int nt = wave; nt < NT; nt += 4) {
    v8f acc = {};
#pragma unroll
    for (int kt = 0; kt < KT; ++kt) {
      v16bf a = load_a_rm(sh_d, COUT, kt * 32);
      v16bf b = load_b_staged(SWsc, (long)kt * NT + nt);
      acc = WMMA_BF16(a, b, acc);
    }
    int col = nt * 16 + col0;
    float bb = bs[col];
#pragma unroll
    for (int r = 0; r < 8; ++r)
      sh_sc[(rb + r) * COUT + col] = sigmf(acc[r] + bb);
  }
  // t1 = hswish(xj@W1 + b1)
  for (int nt = wave; nt < NT1; nt += 4) {
    v8f acc = {};
#pragma unroll
    for (int kt = 0; kt < KT; ++kt) {
      v16bf a = load_a_rm(sh_xj, COUT, kt * 32);
      v16bf b = load_b_staged(SW1, (long)kt * NT1 + nt);
      acc = WMMA_BF16(a, b, acc);
    }
    int col = nt * 16 + col0;
    float bb = b1[col];
#pragma unroll
    for (int r = 0; r < 8; ++r)
      sh_t1[(rb + r) * CQP + col] = (__bf16)hswishf(acc[r] + bb);
  }
  __syncthreads();

  // msg = score * hswish(t1@W2 + b2) ; scatter-add to agg[dst]
  for (int nt = wave; nt < NT; nt += 4) {
    v8f acc = {};
#pragma unroll
    for (int kt = 0; kt < KT2; ++kt) {
      v16bf a = load_a_rm(sh_t1, CQP, kt * 32);
      v16bf b = load_b_staged(SW2, (long)kt * NT + nt);
      acc = WMMA_BF16(a, b, acc);
    }
    int col = nt * 16 + col0;
    float bb = b2[col];
#pragma unroll
    for (int r = 0; r < 8; ++r) {
      int row = rb + r;
      if (sh_ok[row]) {
        float msg = sh_sc[row * COUT + col] * hswishf(acc[r] + bb);
        atomicAdd(&agg[(long)sh_di[row] * COUT + col], msg);
      }
    }
  }
}

// ------------------------------------------------------- grid conv kernels
__global__ void k_scatter_max(const float* __restrict__ xout, const float* __restrict__ xin,
                              const int* __restrict__ batch, int Nn, int G, float s,
                              unsigned* __restrict__ gbuf) {
  long i = (long)blockIdx.x * blockDim.x + threadIdx.x;
  if (i >= (long)Nn * 320) return;
  int c = (int)(i % 320); int n = (int)(i / 320);
  float px = xin[n * 4 + 0] * 256.0f;
  float py = xin[n * 4 + 1] * 256.0f;
  int cx = (int)floorf(px / s); cx = cx < 0 ? 0 : (cx > G - 1 ? G - 1 : cx);
  int cy = (int)floorf(py / s); cy = cy < 0 ? 0 : (cy > G - 1 ? G - 1 : cy);
  long cluster = (long)batch[n] * (G * G) + cy * G + cx;
  atomicMax(&gbuf[cluster * 320 + c], f2ord(xout[i]));
}

// grouped 3x3 VALID conv, decodes grid cells inline (NCHW out, Go=G-2)
__global__ void k_conv3(const unsigned* __restrict__ gbuf, const float* __restrict__ w3,
                        float* __restrict__ out, int G, int Go, long total) {
  long i = (long)blockIdx.x * blockDim.x + threadIdx.x;
  if (i >= total) return;
  int x  = (int)(i % Go);
  int y  = (int)((i / Go) % Go);
  int co = (int)((i / ((long)Go * Go)) % 320);
  int b  = (int)(i / ((long)Go * Go * 320));
  int cb = (co / 40) * 40;
  float s = 0.0f;
  for (int ci = 0; ci < 40; ++ci)
    for (int ky = 0; ky < 3; ++ky)
      for (int kx = 0; kx < 3; ++kx) {
        float v = dec_cell(gbuf[((long)b * G * G + (long)(y + ky) * G + (x + kx)) * 320 + cb + ci]);
        s += v * w3[(((long)co * 40 + ci) * 3 + ky) * 3 + kx];
      }
  out[i] = s;
}

__global__ void k_chstats_nchw(const float* __restrict__ X, int B_, int C, int HW,
                               float* mean, float* istd) {
  int c = blockIdx.x;
  __shared__ float s1[256], s2[256];
  float a = 0.f, b = 0.f;
  long cnt = (long)B_ * HW;
  for (long idx = threadIdx.x; idx < cnt; idx += 256) {
    long bb = idx / HW, p = idx % HW;
    float v = X[(bb * C + c) * HW + p];
    a += v; b += v * v;
  }
  s1[threadIdx.x] = a; s2[threadIdx.x] = b;
  __syncthreads();
  for (int s = 128; s > 0; s >>= 1) {
    if (threadIdx.x < s) { s1[threadIdx.x] += s1[threadIdx.x + s]; s2[threadIdx.x] += s2[threadIdx.x + s]; }
    __syncthreads();
  }
  if (threadIdx.x == 0) {
    float m = s1[0] / cnt;
    float v = s2[0] / cnt - m * m;
    mean[c] = m; istd[c] = rsqrtf(v + 1e-5f);
  }
}
__global__ void k_bn_hswish_nchw(float* __restrict__ X, const float* mean, const float* istd,
                                 const float* gam, const float* bet, int C, int HW, long total) {
  long i = (long)blockIdx.x * blockDim.x + threadIdx.x;
  if (i >= total) return;
  int c = (int)((i / HW) % C);
  X[i] = hswishf((X[i] - mean[c]) * istd[c] * gam[c] + bet[c]);
}

// grouped 1x1 conv (8 groups), NCHW
__global__ void k_conv1x1g(const float* __restrict__ X, const float* __restrict__ w,
                           float* __restrict__ out, int Cin, int Cout, int HW, long total) {
  long i = (long)blockIdx.x * blockDim.x + threadIdx.x;
  if (i >= total) return;
  int p  = (int)(i % HW);
  int co = (int)((i / HW) % Cout);
  int b  = (int)(i / ((long)HW * Cout));
  int cig = Cin / 8;
  int base = (co / (Cout / 8)) * cig;
  float s = 0.0f;
  for (int ci = 0; ci < cig; ++ci)
    s += X[((long)b * Cin + base + ci) * HW + p] * w[(long)co * cig + ci];
  out[i] = s;
}

__global__ void k_spatmean(const float* __restrict__ T, float* out, int C, int HW) {
  int bc = blockIdx.x;                 // b*C + c
  __shared__ float s1[256];
  float a = 0.f;
  for (int p = threadIdx.x; p < HW; p += 256) a += T[(long)bc * HW + p];
  s1[threadIdx.x] = a; __syncthreads();
  for (int s = 128; s > 0; s >>= 1) {
    if (threadIdx.x < s) s1[threadIdx.x] += s1[threadIdx.x + s];
    __syncthreads();
  }
  if (threadIdx.x == 0) out[bc] = s1[0] / HW;
}

__global__ void k_ss(const float* __restrict__ T, const float* __restrict__ tmean,
                     const float* __restrict__ wss, const float* __restrict__ bss,
                     float* __restrict__ SS, int HW, long total) {
  long i = (long)blockIdx.x * blockDim.x + threadIdx.x;
  if (i >= total) return;
  int p = (int)(i % HW); int b = (int)(i / HW);
  float s = bss[0];
  for (int c = 0; c < 80; ++c)
    s += T[((long)b * 80 + c) * HW + p] * wss[c] + tmean[b * 80 + c] * wss[80 + c];
  SS[i] = sigmf(s);
}

__global__ void k_xavg(const float* __restrict__ H2, const float* __restrict__ SS,
                       float* __restrict__ xavg, int C, int HW) {
  int bc = blockIdx.x;                 // b*C + c
  int b = bc / C;
  __shared__ float s1[256];
  float a = 0.f;
  for (int p = threadIdx.x; p < HW; p += 256)
    a += H2[(long)bc * HW + p] * SS[(long)b * HW + p];
  s1[threadIdx.x] = a; __syncthreads();
  for (int s = 128; s > 0; s >>= 1) {
    if (threadIdx.x < s) s1[threadIdx.x] += s1[threadIdx.x + s];
    __syncthreads();
  }
  if (threadIdx.x == 0) xavg[bc] = s1[0] / HW;
}

__global__ void k_cs(const float* __restrict__ xavg, const float* __restrict__ wcs,
                     const float* __restrict__ bcs, float* __restrict__ xcat, int gidx, long total) {
  long i = (long)blockIdx.x * blockDim.x + threadIdx.x;
  if (i >= total) return;
  int c = (int)(i % 320); int b = (int)(i / 320);
  int base = (c / 40) * 40;
  float s = bcs[c];
  for (int k = 0; k < 40; ++k) s += xavg[b * 320 + base + k] * wcs[c * 40 + k];
  xcat[(long)b * 960 + gidx * 320 + c] = sigmf(s) * xavg[i];
}

__global__ void k_fc2(const float* __restrict__ xcat, const float* __restrict__ w,
                      float* __restrict__ y2, long total) {
  long i = (long)blockIdx.x * blockDim.x + threadIdx.x;
  if (i >= total) return;
  int o = (int)(i % 1280); int b = (int)(i / 1280);
  int base = (o / 160) * 120;
  float s = 0.0f;
  for (int k = 0; k < 120; ++k) s += xcat[(long)b * 960 + base + k] * w[(long)o * 120 + k];
  y2[i] = s;
}

__global__ void k_bn_feat(const float* __restrict__ y2, const float* mean, const float* istd,
                          const float* gam, const float* bet,
                          float* __restrict__ feat, float* __restrict__ outfeat, long total) {
  long i = (long)blockIdx.x * blockDim.x + threadIdx.x;
  if (i >= total) return;
  int c = (int)(i % 1280);
  float v = hswishf((y2[i] - mean[c]) * istd[c] * gam[c] + bet[c]);
  feat[i] = v; outfeat[i] = v;
}

__global__ void k_logits(const float* __restrict__ feat, const float* __restrict__ wfc,
                         const float* __restrict__ bfc, float* __restrict__ out, long total) {
  long i = (long)blockIdx.x * blockDim.x + threadIdx.x;
  if (i >= total) return;
  int o = (int)(i % 345); int b = (int)(i / 345);
  float s = bfc[o];
  for (int k = 0; k < 1280; ++k) s += feat[(long)b * 1280 + k] * wfc[(long)k * 345 + o];
  out[i] = s;
}

// ================================================================= launch
extern "C" void kernel_launch(void* const* d_in, const int* in_sizes, int n_in,
                              void* d_out, int out_size, void* d_ws, size_t ws_size,
                              hipStream_t stream) {
  const int N = in_sizes[2];          // 32768
  const int E = in_sizes[1] / 2;      // 524288
  const int Bb = 128;

  static const int CIN[5]   = {4, 68, 196, 388, 644};
  static const int COUTA[5] = {64, 128, 192, 256, 320};
  static const int HIN[5]   = {640, 576, 448, 256, 0};
  static const int HOUT[5]  = {576, 448, 256, 0, -1};
  static const int BIDX[5]  = {4, 14, 23, 32, 41};

  auto F = [&](int i) { return (const float*)d_in[i]; };
  const float* xin   = F(0);
  const int*   esrc  = (const int*)d_in[1];
  const int*   edst  = esrc + E;
  const int*   batch = (const int*)d_in[2];
  const float* bfc     = F(3);
  const float* bn2fc_b = F(50);
  const float* bn2fc_g = F(51);
  const float* w2fc    = F(91);
  const float* wfc     = F(92);

  // -------- workspace layout
  char* ws = (char*)d_ws;
  size_t cur = 0;
  auto alloc = [&](size_t bytes) { size_t r = cur; cur = (cur + bytes + 255) & ~(size_t)255; return r; };
  size_t oHIST = alloc((size_t)N * 644 * 2 + 4096);   // + overrun pad
  size_t oHF   = alloc((size_t)N * 320 * 4);
  size_t oHB   = alloc((size_t)N * 320 * 2);
  size_t oY    = alloc((size_t)N * 320 * 4);
  size_t oAGG  = alloc((size_t)N * 320 * 4);
  size_t oXOUT = alloc((size_t)N * 320 * 4);
  size_t oMEAN = alloc(2048 * 4);
  size_t oISTD = alloc(2048 * 4);
  size_t oWB   = alloc(1u << 22);
  size_t oXCAT = alloc((size_t)Bb * 960 * 4);
  size_t oFEAT = alloc((size_t)Bb * 1280 * 4);
  size_t oY2   = alloc((size_t)Bb * 1280 * 4);
  size_t oTME  = alloc((size_t)Bb * 80 * 4);
  size_t oXAV  = alloc((size_t)Bb * 320 * 4);
  size_t oGB   = alloc((size_t)Bb * 1024 * 320 * 4);
  size_t oH1   = alloc((size_t)Bb * 320 * 900 * 4);
  size_t oH2   = alloc((size_t)Bb * 320 * 900 * 4);
  size_t oT    = alloc((size_t)Bb * 80 * 900 * 4);
  size_t oSS   = alloc((size_t)Bb * 900 * 4);

  __bf16*   HIST = (__bf16*)(ws + oHIST);
  float*    HF   = (float*)(ws + oHF);
  __bf16*   HB   = (__bf16*)(ws + oHB);
  float*    Y    = (float*)(ws + oY);
  float*    AGG  = (float*)(ws + oAGG);
  float*    XOUT = (float*)(ws + oXOUT);
  float*    MEAN = (float*)(ws + oMEAN);
  float*    ISTD = (float*)(ws + oISTD);
  __bf16*   WB   = (__bf16*)(ws + oWB);
  float*    XCAT = (float*)(ws + oXCAT);
  float*    FEAT = (float*)(ws + oFEAT);
  float*    Y2   = (float*)(ws + oY2);
  float*    TME  = (float*)(ws + oTME);
  float*    XAV  = (float*)(ws + oXAV);
  unsigned* GB   = (unsigned*)(ws + oGB);
  float*    H1   = (float*)(ws + oH1);
  float*    H2   = (float*)(ws + oH2);
  float*    Tb   = (float*)(ws + oT);
  float*    SSb  = (float*)(ws + oSS);
  float*    dout = (float*)d_out;

  // -------- staged (fragment-layout) bf16 weight offsets
  size_t off_w[5], off_ws[5], off_w1[5], off_w2[5], wb = 0;
  int KTG[5], CQP[5];
  for (int k = 0; k < 5; ++k) {
    int cout = COUTA[k], cin = CIN[k], g = (k == 0) ? 1 : 4;
    int Kg = cin / g, CQ = cout / 4;
    KTG[k] = CDIV(Kg, 32);
    CQP[k] = CDIV(CQ, 32) * 32;
    off_w[k]  = wb; if (g > 1) wb += (size_t)KTG[k] * 32 * cout;   // padded
    off_ws[k] = wb; wb += (size_t)cout * cout;
    off_w1[k] = wb; wb += (size_t)cout * CQ;
    off_w2[k] = wb; wb += (size_t)CQP[k] * cout;                   // padded
  }

  // -------- per-block param pointers + weight staging
  const float *p_b1[5], *p_b2[5], *p_bnb[5], *p_bng[5], *p_bs[5], *p_w[5], *p_w1[5], *p_w2[5], *p_ws[5];
  const float* p_b0 = nullptr;
  for (int k = 0; k < 5; ++k) {
    int base = BIDX[k];
    if (k == 0) {
      p_b0 = F(base + 0); p_b1[k] = F(base + 1); p_b2[k] = F(base + 2);
      p_bnb[k] = F(base + 3); p_bng[k] = F(base + 4); p_bs[k] = F(base + 5);
      p_w[k] = F(base + 6); p_w1[k] = F(base + 7); p_w2[k] = F(base + 8); p_ws[k] = F(base + 9);
    } else {
      p_b1[k] = F(base + 0); p_b2[k] = F(base + 1); p_bnb[k] = F(base + 2);
      p_bng[k] = F(base + 3); p_bs[k] = F(base + 4); p_w[k] = F(base + 5);
      p_w1[k] = F(base + 6); p_w2[k] = F(base + 7); p_ws[k] = F(base + 8);
    }
    int cout = COUTA[k], CQ = cout / 4;
    if (k > 0) {
      long tot = (long)KTG[k] * 32 * cout;
      k_stage_nodeW<<<CDIV(tot, 256), 256, 0, stream>>>(p_w[k], WB + off_w[k],
                                                        CIN[k] / 4, cout / 4, cout, tot);
    }
    long tws = (long)cout * cout;
    k_stage_kn<<<CDIV(tws, 256), 256, 0, stream>>>(p_ws[k], WB + off_ws[k], cout, cout, tws);
    long tw1 = (long)cout * CQ;
    k_stage_kn<<<CDIV(tw1, 256), 256, 0, stream>>>(p_w1[k], WB + off_w1[k], cout, CQ, tw1);
    long tw2 = (long)CQP[k] * cout;
    k_stage_kn<<<CDIV(tw2, 256), 256, 0, stream>>>(p_w2[k], WB + off_w2[k], CQ, cout, tw2);
  }

  // zero-init hist (+pad) so any K-overrun A reads are finite, then stage x
  k_fill_u32<<<CDIV((long)N * 322 + 1024, 256), 256, 0, stream>>>((unsigned*)HIST, 0u,
                                                                  (long)N * 322 + 1024);
  k_x_to_hist<<<CDIV((long)N * 4, 256), 256, 0, stream>>>(xin, HIST, N);

  // -------- GEC blocks
  const long E2 = (long)E + N;
  for (int k = 0; k < 5; ++k) {
    int cout = COUTA[k], g = (k == 0) ? 1 : 4;
    int Kg = CIN[k] / g, cgo = cout / g;
    long NC = (long)N * cout;

    if (k == 0) {
      k_node0<<<CDIV((long)N * 64, 256), 256, 0, stream>>>(xin, p_w[0], p_b0, Y, N);
    } else {
      dim3 gr(N / 16, cout / 16);
      k_node_gemm_g<<<gr, 32, 0, stream>>>(HIST, HIN[k], WB + off_w[k], cout, Kg, cgo, KTG[k], Y);
    }
    k_colstats<<<cout, 256, 0, stream>>>(Y, N, cout, MEAN, ISTD);
    k_bn_apply_node<<<CDIV(NC, 256), 256, 0, stream>>>(Y, MEAN, ISTD, p_bng[k], p_bnb[k],
                                                       cout, g, cgo, HF, HB, NC);
    k_zero_f32<<<CDIV(NC, 256), 256, 0, stream>>>(AGG, NC);

    dim3 eg((unsigned)CDIV(E2, 16));
    switch (cout) {
      case 64:  k_edge<64><<<eg, 128, 0, stream>>>(HB, esrc, edst, E, N, WB + off_ws[k], p_bs[k], WB + off_w1[k], p_b1[k], WB + off_w2[k], p_b2[k], AGG); break;
      case 128: k_edge<128><<<eg, 128, 0, stream>>>(HB, esrc, edst, E, N, WB + off_ws[k], p_bs[k], WB + off_w1[k], p_b1[k], WB + off_w2[k], p_b2[k], AGG); break;
      case 192: k_edge<192><<<eg, 128, 0, stream>>>(HB, esrc, edst, E, N, WB + off_ws[k], p_bs[k], WB + off_w1[k], p_b1[k], WB + off_w2[k], p_b2[k], AGG); break;
      case 256: k_edge<256><<<eg, 128, 0, stream>>>(HB, esrc, edst, E, N, WB + off_ws[k], p_bs[k], WB + off_w1[k], p_b1[k], WB + off_w2[k], p_b2[k], AGG); break;
      default:  k_edge<320><<<eg, 128, 0, stream>>>(HB, esrc, edst, E, N, WB + off_ws[k], p_bs[k], WB + off_w1[k], p_b1[k], WB + off_w2[k], p_b2[k], AGG); break;
    }
    k_finalize<<<CDIV(NC, 256), 256, 0, stream>>>(HF, AGG, cout,
                                                  (k < 4) ? HIST : nullptr, (k < 4) ? HOUT[k] : 0,
                                                  (k == 4) ? XOUT : nullptr, NC);
  }

  // -------- grid conv path
  static const int GS[3] = {8, 16, 32};
  for (int gi = 0; gi < 3; ++gi) {
    int G = GS[gi];
    float sc = (float)(256 / G);       // ceil(256/G) exact
    int Go = G - 2, HW = Go * Go;
    long cells = (long)Bb * G * G * 320;
    k_fill_u32<<<CDIV(cells, 256), 256, 0, stream>>>(GB, 0x007FFFFFu, cells); // ord(-inf)
    k_scatter_max<<<CDIV((long)N * 320, 256), 256, 0, stream>>>(XOUT, xin, batch, N, G, sc, GB);

    int qb = 52 + 13 * gi;
    const float* bcs   = F(qb + 0);
    const float* bn1_b = F(qb + 1); const float* bn1_g = F(qb + 2);
    const float* bn2_b = F(qb + 3); const float* bn2_g = F(qb + 4);
    const float* bst_b = F(qb + 5); const float* bst_g = F(qb + 6);
    const float* bss   = F(qb + 7);
    const float* w1x1  = F(qb + 8); const float* w3 = F(qb + 9);
    const float* wcs   = F(qb + 10); const float* wss = F(qb + 11);
    const float* wst   = F(qb + 12);

    long t1 = (long)Bb * 320 * HW;
    k_conv3<<<CDIV(t1, 256), 256, 0, stream>>>(GB, w3, H1, G, Go, t1);
    k_chstats_nchw<<<320, 256, 0, stream>>>(H1, Bb, 320, HW, MEAN, ISTD);
    k_bn_hswish_nchw<<<CDIV(t1, 256), 256, 0, stream>>>(H1, MEAN, ISTD, bn1_g, bn1_b, 320, HW, t1);

    k_conv1x1g<<<CDIV(t1, 256), 256, 0, stream>>>(H1, w1x1, H2, 320, 320, HW, t1);
    k_chstats_nchw<<<320, 256, 0, stream>>>(H2, Bb, 320, HW, MEAN, ISTD);
    k_bn_hswish_nchw<<<CDIV(t1, 256), 256, 0, stream>>>(H2, MEAN, ISTD, bn2_g, bn2_b, 320, HW, t1);

    long t2 = (long)Bb * 80 * HW;
    k_conv1x1g<<<CDIV(t2, 256), 256, 0, stream>>>(H2, wst, Tb, 320, 80, HW, t2);
    k_chstats_nchw<<<80, 256, 0, stream>>>(Tb, Bb, 80, HW, MEAN, ISTD);
    k_bn_hswish_nchw<<<CDIV(t2, 256), 256, 0, stream>>>(Tb, MEAN, ISTD, bst_g, bst_b, 80, HW, t2);

    k_spatmean<<<Bb * 80, 256, 0, stream>>>(Tb, TME, 80, HW);
    long t3 = (long)Bb * HW;
    k_ss<<<CDIV(t3, 256), 256, 0, stream>>>(Tb, TME, wss, bss, SSb, HW, t3);
    k_xavg<<<Bb * 320, 256, 0, stream>>>(H2, SSb, XAV, 320, HW);
    k_cs<<<CDIV((long)Bb * 320, 256), 256, 0, stream>>>(XAV, wcs, bcs, XCAT, gi, (long)Bb * 320);
  }

  // -------- final FC head
  long tf = (long)Bb * 1280;
  k_fc2<<<CDIV(tf, 256), 256, 0, stream>>>(XCAT, w2fc, Y2, tf);
  k_colstats<<<1280, 256, 0, stream>>>(Y2, Bb, 1280, MEAN, ISTD);
  k_bn_feat<<<CDIV(tf, 256), 256, 0, stream>>>(Y2, MEAN, ISTD, bn2fc_g, bn2fc_b,
                                               FEAT, dout + (long)Bb * 345, tf);
  k_logits<<<CDIV((long)Bb * 345, 256), 256, 0, stream>>>(FEAT, wfc, bfc, dout, (long)Bb * 345);
  (void)n_in; (void)out_size; (void)ws_size;
}
